// SimpleAttention_11192684774195
// MI455X (gfx1250) — compile-verified, matrix operand order checked by reading, not run on hardware
//
#include <hip/hip_runtime.h>
#include <hip/hip_bf16.h>

#define H   1024
#define NHD 16
#define HD  64
#define BB  2
#define SS  2048
#define NT  (BB*SS)     // 4096 tokens
#define H3  (3*H)       // 3072
#define LDB 48          // padded LDS row stride (halves): 96B, 16B aligned,
                        // breaks 64-bank conflicts vs a 64B stride

typedef __attribute__((ext_vector_type(4)))  _Float16 v4h;
typedef __attribute__((ext_vector_type(8)))  _Float16 v8h;
typedef __attribute__((ext_vector_type(16))) _Float16 v16h;
typedef __attribute__((ext_vector_type(8)))  float    v8f;

union U16H { v16h v; v8h h[2]; };

// Load a 16x32 f16 A-fragment (or symmetric B-fragment, where "row" = output
// column) from a row-major matrix with leading dimension ld (in halves).
// Lane mapping per CDNA5 ISA 7.12.2: lanes 0-15 row m=lane, K = {0..7,16..23};
// lanes 16-31 row m=lane-16, K = {8..15,24..31}. Two contiguous 16B loads.
__device__ __forceinline__ v16h frag_ld(const _Float16* base, int ld, int k0, int lane) {
  const int r  = lane & 15;
  const int hb = (lane & 16) ? 8 : 0;
  const _Float16* p = base + (size_t)r * ld + k0 + hb;
  U16H u;
  u.h[0] = *(const v8h*)(p);
  u.h[1] = *(const v8h*)(p + 16);
  return u.v;
}

__device__ __forceinline__ v8f wmma32(v16h a, v16h b, v8f c) {
  return __builtin_amdgcn_wmma_f32_16x16x32_f16(false, a, false, b, (short)0, c, false, false);
}

// LDS byte offset of a generic pointer to __shared__ (flat aperture mapping:
// LDS offset is the low 32 bits of the flat address, ISA 10.2).
__device__ __forceinline__ unsigned lds_off(const _Float16* p) {
  return (unsigned)(unsigned long long)p;
}

// Async 16B global -> LDS copy (GLOBAL_LOAD_ASYNC_TO_LDS_B128, ASYNCcnt).
__device__ __forceinline__ void async_ld16(unsigned ldsoff, const _Float16* gptr) {
  asm volatile("global_load_async_to_lds_b128 %0, %1, off"
               :: "v"(ldsoff), "v"((unsigned long long)(size_t)gptr)
               : "memory");
}

__device__ __forceinline__ void wait_async0() {
  asm volatile("s_wait_asynccnt 0x0" ::: "memory");
}

// ---------------------------------------------------------------- cast kernel
__global__ void __launch_bounds__(256) cast_f32_f16(const float4* __restrict__ s,
                                                    v4h* __restrict__ d, int n4) {
  int i  = blockIdx.x * blockDim.x + threadIdx.x;
  int st = gridDim.x * blockDim.x;
  for (; i < n4; i += st) {
    float4 f = s[i];
    v4h o;
    o[0] = (_Float16)f.x; o[1] = (_Float16)f.y;
    o[2] = (_Float16)f.z; o[3] = (_Float16)f.w;
    d[i] = o;
  }
}

// --------------------------------------------------------------- GEMM engine
// Block = 128 threads = 4 waves, block tile 32(M) x 128(N), wave tile 16x64.
// Per 32-deep K-step: A(32x32)+B(128x32) slices are copied global->LDS with
// async-to-LDS (no data VGPRs, ASYNCcnt), double-buffered. One
// s_wait_asynccnt + one barrier per step: the wait covers this wave's copies,
// the barrier publishes completion block-wide AND orders last step's reads of
// the buffer we are about to overwrite.
struct GemmCtx {
  v8f  acc[4];
  int  lane, wv, m0w, n0w;           // per-wave tile origin
};

template <typename EPILOG>
__device__ __forceinline__ void gemm_body(const _Float16* __restrict__ A,
                                          const _Float16* __restrict__ W,
                                          int m0, int n0,
                                          _Float16* __restrict__ lA,  // [2][32*LDB]
                                          _Float16* __restrict__ lB,  // [2][128*LDB]
                                          EPILOG epilog) {
  const int tid  = threadIdx.x;
  const int lane = tid & 31;
  const int wv   = tid >> 5;
  const int mi   = wv >> 1;          // 0..1
  const int ni   = wv & 1;           // 0..1

  // Staging assignment: 8-half (16B) chunks per lane.
  const int arow = tid >> 2;         // 0..31
  const int achk = (tid & 3) * 8;    // 0,8,16,24
  const int brow = tid >> 2;         // + {0,32,64,96}
  const int bchk = (tid & 3) * 8;

  const _Float16* ga  = A + (size_t)(m0 + arow) * H + achk;
  const _Float16* gb0 = W + (size_t)(n0 + brow +  0) * H + bchk;
  const _Float16* gb1 = W + (size_t)(n0 + brow + 32) * H + bchk;
  const _Float16* gb2 = W + (size_t)(n0 + brow + 64) * H + bchk;
  const _Float16* gb3 = W + (size_t)(n0 + brow + 96) * H + bchk;

  unsigned aoff[2], boff[2];
  #pragma unroll
  for (int p = 0; p < 2; ++p) {
    aoff[p] = lds_off(lA + (size_t)p * (32 * LDB)  + arow * LDB + achk);
    boff[p] = lds_off(lB + (size_t)p * (128 * LDB) + brow * LDB + bchk);
  }
  const unsigned bstep = 32 * LDB * 2;   // 32 rows, bytes

  v8f acc[4] = {};

  // Prologue: slice 0 into buffer 0.
  async_ld16(aoff[0], ga);
  async_ld16(boff[0] + 0 * bstep, gb0);
  async_ld16(boff[0] + 1 * bstep, gb1);
  async_ld16(boff[0] + 2 * bstep, gb2);
  async_ld16(boff[0] + 3 * bstep, gb3);

  for (int kk = 0; kk < H / 32; ++kk) {
    const int p = kk & 1;
    wait_async0();       // this wave's copies for slice kk are complete
    __syncthreads();     // block-wide: buf[p] ready; buf[p^1] reads retired

    // Issue async copies for slice kk+1 into the other buffer (clamped dummy
    // on the last step; s_endpgm's implicit wait-idle drains it).
    int kn = (kk + 1) * 32;
    if (kn > H - 32) kn = H - 32;
    async_ld16(aoff[p ^ 1], ga + kn);
    async_ld16(boff[p ^ 1] + 0 * bstep, gb0 + kn);
    async_ld16(boff[p ^ 1] + 1 * bstep, gb1 + kn);
    async_ld16(boff[p ^ 1] + 2 * bstep, gb2 + kn);
    async_ld16(boff[p ^ 1] + 3 * bstep, gb3 + kn);

    // Compute phase off buf[p]: fragments from LDS, 4 chained WMMAs per wave.
    const _Float16* lAp = lA + (size_t)p * (32 * LDB);
    const _Float16* lBp = lB + (size_t)p * (128 * LDB);
    v16h af = frag_ld(lAp + (size_t)(mi * 16) * LDB, LDB, 0, lane);
    #pragma unroll
    for (int nt = 0; nt < 4; ++nt) {
      v16h bf = frag_ld(lBp + (size_t)(ni * 64 + nt * 16) * LDB, LDB, 0, lane);
      acc[nt] = wmma32(af, bf, acc[nt]);
    }
  }

  GemmCtx c;
  c.acc[0] = acc[0]; c.acc[1] = acc[1]; c.acc[2] = acc[2]; c.acc[3] = acc[3];
  c.lane = lane; c.wv = wv;
  c.m0w = m0 + mi * 16; c.n0w = n0 + ni * 64;
  epilog(c);
}

// ------------------------------------------------------------- QKV projection
// C[4096,3072] = Xh * Wqkv^T, scattered into qkvh [3][B][NH][S][HD] as f16.
__global__ void __launch_bounds__(128) gemm_qkv(const _Float16* __restrict__ xh,
                                                const _Float16* __restrict__ wh,
                                                _Float16* __restrict__ qkvh) {
  __shared__ __align__(16) _Float16 lA[2 * 32 * LDB];
  __shared__ __align__(16) _Float16 lB[2 * 128 * LDB];
  const int bm = blockIdx.x / 24;   // 0..127
  const int bn = blockIdx.x % 24;   // 0..23
  gemm_body(xh, wh, bm * 32, bn * 128, lA, lB, [=](const GemmCtx& c) {
    const int n  = c.lane & 15;
    const int mb = (c.lane & 16) ? 8 : 0;
    #pragma unroll
    for (int nt = 0; nt < 4; ++nt) {
      const int o    = c.n0w + nt * 16 + n;
      const int part = o >> 10;          // 0..2 -> q/k/v
      const int oin  = o & 1023;
      const int head = oin >> 6;
      const int dd   = oin & 63;
      #pragma unroll
      for (int r = 0; r < 8; ++r) {
        const int tok = c.m0w + mb + r;
        const int bb  = tok >> 11;
        const int ss  = tok & 2047;
        qkvh[((((size_t)part * BB + bb) * NHD + head) * SS + ss) * HD + dd] =
            (_Float16)c.acc[nt][r];
      }
    }
  });
}

// ------------------------------------------------------------ out projection
// out[4096,1024] = Ah * Wout^T, fp32 result.
__global__ void __launch_bounds__(128) gemm_out(const _Float16* __restrict__ ah,
                                                const _Float16* __restrict__ wh,
                                                float* __restrict__ out) {
  __shared__ __align__(16) _Float16 lA[2 * 32 * LDB];
  __shared__ __align__(16) _Float16 lB[2 * 128 * LDB];
  const int bm = blockIdx.x >> 3;   // 0..127
  const int bn = blockIdx.x & 7;    // 0..7
  gemm_body(ah, wh, bm * 32, bn * 128, lA, lB, [=](const GemmCtx& c) {
    const int n  = c.lane & 15;
    const int mb = (c.lane & 16) ? 8 : 0;
    #pragma unroll
    for (int nt = 0; nt < 4; ++nt) {
      const int o = c.n0w + nt * 16 + n;
      #pragma unroll
      for (int r = 0; r < 8; ++r)
        out[(size_t)(c.m0w + mb + r) * H + o] = c.acc[nt][r];
    }
  });
}

// ------------------------------------------------------------ flash attention
// One wave per (b, head, 16-query tile). 32-key steps, online softmax.
// K fragments and the V tile for step s+1 are prefetched into registers at
// the top of step s, hiding global latency behind softmax + PV WMMAs.
__global__ void __launch_bounds__(128) attn_fwd(const _Float16* __restrict__ qkvh,
                                                _Float16* __restrict__ attnh) {
  __shared__ __align__(16) _Float16 Pt[4][16 * 32];   // probs, per wave
  __shared__ __align__(16) _Float16 Vt[4][64 * 32];   // V transposed, per wave
  const int lane = threadIdx.x & 31;
  const int wv   = threadIdx.x >> 5;
  const int t    = blockIdx.x * 4 + wv;               // 4096 tasks
  const int b    = t >> 11;
  const int rem  = t & 2047;
  const int h    = rem >> 7;
  const int q0   = (rem & 127) << 4;

  const _Float16* Qb = qkvh + (((size_t)(0 * BB + b) * NHD + h) * SS) * HD;
  const _Float16* Kb = qkvh + (((size_t)(1 * BB + b) * NHD + h) * SS) * HD;
  const _Float16* Vb = qkvh + (((size_t)(2 * BB + b) * NHD + h) * SS) * HD;

  const v16h qf0 = frag_ld(Qb + (size_t)q0 * HD, HD, 0,  lane);
  const v16h qf1 = frag_ld(Qb + (size_t)q0 * HD, HD, 32, lane);

  v8f   acc[4] = {};
  float mrow[8], lrow[8];
  #pragma unroll
  for (int r = 0; r < 8; ++r) { mrow[r] = -3.0e38f; lrow[r] = 0.0f; }

  const int   n      = lane & 15;
  const int   mb     = (lane & 16) ? 8 : 0;
  const float scale  = 0.125f;                 // 1/sqrt(64)
  const int   nsteps = (q0 + 47) >> 5;

  // Prologue: fragments for step 0.
  v16h kf0 = frag_ld(Kb,           HD, 0,  lane);
  v16h kf1 = frag_ld(Kb,           HD, 32, lane);
  v16h kf2 = frag_ld(Kb + 16 * HD, HD, 0,  lane);
  v16h kf3 = frag_ld(Kb + 16 * HD, HD, 32, lane);
  v8h  vreg[8];
  {
    const v8h* vp = (const v8h*)(Vb + (size_t)lane * HD);
    #pragma unroll
    for (int c8 = 0; c8 < 8; ++c8) vreg[c8] = vp[c8];
  }

  for (int stp = 0; stp < nsteps; ++stp) {
    const int kbase = stp << 5;
    int knext = kbase + 32;
    if (knext > SS - 32) knext = SS - 32;      // clamp: dummy (unused) prefetch

    // Prefetch step s+1 K fragments and V tile into registers.
    v16h nk0 = frag_ld(Kb + (size_t)knext * HD,        HD, 0,  lane);
    v16h nk1 = frag_ld(Kb + (size_t)knext * HD,        HD, 32, lane);
    v16h nk2 = frag_ld(Kb + (size_t)(knext + 16) * HD, HD, 0,  lane);
    v16h nk3 = frag_ld(Kb + (size_t)(knext + 16) * HD, HD, 32, lane);
    v8h  nv[8];
    {
      const v8h* vp = (const v8h*)(Vb + (size_t)(knext + lane) * HD);
      #pragma unroll
      for (int c8 = 0; c8 < 8; ++c8) nv[c8] = vp[c8];
    }

    // Stage current V tile transposed into LDS: Vt[d][key].
    #pragma unroll
    for (int c8 = 0; c8 < 8; ++c8)
      #pragma unroll
      for (int e = 0; e < 8; ++e)
        Vt[wv][(c8 * 8 + e) * 32 + lane] = vreg[c8][e];

    // Scores: registers only -> no load wait in front of these WMMAs.
    v8f s0 = {}, s1 = {};
    s0 = wmma32(qf0, kf0, s0);
    s0 = wmma32(qf1, kf1, s0);
    s1 = wmma32(qf0, kf2, s1);
    s1 = wmma32(qf1, kf3, s1);

    // Causal mask + online softmax. Row m lives on 16 lanes of one half;
    // butterfly xor masks 1,2,4,8 stay inside the half (wave32).
    #pragma unroll
    for (int r = 0; r < 8; ++r) {
      const int qrow = q0 + mb + r;
      float a0 = s0[r] * scale;
      float a1 = s1[r] * scale;
      if (kbase + n      > qrow) a0 = -3.0e38f;
      if (kbase + 16 + n > qrow) a1 = -3.0e38f;
      float mx = fmaxf(a0, a1);
      mx = fmaxf(mx, __shfl_xor(mx, 1, 32));
      mx = fmaxf(mx, __shfl_xor(mx, 2, 32));
      mx = fmaxf(mx, __shfl_xor(mx, 4, 32));
      mx = fmaxf(mx, __shfl_xor(mx, 8, 32));
      const float mnew  = fmaxf(mrow[r], mx);
      const float alpha = __expf(mrow[r] - mnew);
      const float p0 = __expf(a0 - mnew);
      const float p1 = __expf(a1 - mnew);
      float sm = p0 + p1;
      sm += __shfl_xor(sm, 1, 32);
      sm += __shfl_xor(sm, 2, 32);
      sm += __shfl_xor(sm, 4, 32);
      sm += __shfl_xor(sm, 8, 32);
      lrow[r] = lrow[r] * alpha + sm;
      mrow[r] = mnew;
      acc[0][r] *= alpha; acc[1][r] *= alpha;
      acc[2][r] *= alpha; acc[3][r] *= alpha;
      Pt[wv][(mb + r) * 32 + n]      = (_Float16)p0;
      Pt[wv][(mb + r) * 32 + 16 + n] = (_Float16)p1;
    }

    // Intra-wave LDS RAW: DS ops are in-order per wave, one wait covers
    // both the P stores and the Vt transpose stores.
    asm volatile("s_wait_dscnt 0" ::: "memory");

    const v16h pf = frag_ld(&Pt[wv][0], 32, 0, lane);
    #pragma unroll
    for (int dseg = 0; dseg < 4; ++dseg) {
      v16h vf = frag_ld(&Vt[wv][0] + (size_t)(dseg * 16) * 32, 32, 0, lane);
      acc[dseg] = wmma32(pf, vf, acc[dseg]);
    }

    // Rotate double buffers.
    kf0 = nk0; kf1 = nk1; kf2 = nk2; kf3 = nk3;
    #pragma unroll
    for (int c8 = 0; c8 < 8; ++c8) vreg[c8] = nv[c8];
  }

  // Normalize and store f16 [B,S,H].
  #pragma unroll
  for (int r = 0; r < 8; ++r) {
    const float  inv = 1.0f / lrow[r];
    const size_t row = (size_t)(b * SS + q0 + mb + r) * H + h * HD;
    #pragma unroll
    for (int dseg = 0; dseg < 4; ++dseg)
      attnh[row + dseg * 16 + n] = (_Float16)(acc[dseg][r] * inv);
  }
}

// ------------------------------------------------------------------ launcher
extern "C" void kernel_launch(void* const* d_in, const int* in_sizes, int n_in,
                              void* d_out, int out_size, void* d_ws, size_t ws_size,
                              hipStream_t stream) {
  const float* x     = (const float*)d_in[0];   // [2,2048,1024]
  const float* w_qkv = (const float*)d_in[1];   // [3072,1024]
  const float* w_out = (const float*)d_in[2];   // [1024,1024]
  float*       outp  = (float*)d_out;           // [2,2048,1024]

  char* ws = (char*)d_ws;
  _Float16* xh    = (_Float16*)(ws);                         //  8 MB
  _Float16* wqh   = (_Float16*)(ws + 8388608);               //  6 MB
  _Float16* woh   = (_Float16*)(ws + 14680064);              //  2 MB
  _Float16* qkvh  = (_Float16*)(ws + 16777216);              // 24 MB [3][B][NH][S][HD]
  _Float16* attnh = (_Float16*)(ws + 41943040);              //  8 MB [B][S][H]

  cast_f32_f16<<<1024, 256, 0, stream>>>((const float4*)x,     (v4h*)xh,  (NT * H) / 4);
  cast_f32_f16<<<768,  256, 0, stream>>>((const float4*)w_qkv, (v4h*)wqh, (H3 * H) / 4);
  cast_f32_f16<<<256,  256, 0, stream>>>((const float4*)w_out, (v4h*)woh, (H * H) / 4);

  gemm_qkv<<<3072, 128, 0, stream>>>(xh, wqh, qkvh);    // 32x128 block tiles
  attn_fwd<<<1024, 128, 0, stream>>>(qkvh, attnh);      // 4096 waves
  gemm_out<<<1024, 128, 0, stream>>>(attnh, woh, outp); // 32x128 block tiles
  (void)in_sizes; (void)n_in; (void)out_size; (void)ws_size;
}
